// MultiHeadAttention_29145648071250
// MI455X (gfx1250) — compile-verified, matrix operand order checked by reading, not run on hardware
//
#include <hip/hip_runtime.h>
#include <hip/hip_bf16.h>

// ---------------------------------------------------------------------------
// MI455X (gfx1250) multi-head attention forward.
// Memory-bound on the 537MB attn output => write attn exactly once, keep the
// 16x2048 logit panel resident in LDS (320KB/WGP on CDNA5), bf16 WMMA
// (v_wmma_f32_16x16x32_bf16) for all GEMMs with f32 accumulation.
// Weights are pre-converted to bf16 once (they are reused by every tile);
// activations convert in-register on first touch.
// ---------------------------------------------------------------------------

typedef __attribute__((ext_vector_type(16))) __bf16 v16bf;
typedef __attribute__((ext_vector_type(8)))  __bf16 v8bf;
typedef __attribute__((ext_vector_type(8)))  float  v8f;
typedef __attribute__((ext_vector_type(4)))  float  v4f;

#define MDIM   1024
#define NHEADS 16
#define DEPTHD 64
#define BATCH  2
#define SEQ    2048
#define PST    2052   // padded LDS row stride (floats): 2052 % 64 == 4 -> no bank alignment

// --- fragment builders -------------------------------------------------------
// A (16x32 MxK) and B-as-[N,K]-row-major share the same per-lane pattern:
// lane L: row = L%16, kgrp = L/16; elems 0..7 at k = kgrp*8 + 0..7,
// elems 8..15 at k = 16 + kgrp*8 + 0..7 (two 16-byte chunks).
static __device__ __forceinline__ v16bf frag_from_bf16(const __bf16* row, int kgrp) {
  v8bf lo = *reinterpret_cast<const v8bf*>(row + kgrp * 8);
  v8bf hi = *reinterpret_cast<const v8bf*>(row + 16 + kgrp * 8);
  return __builtin_shufflevector(lo, hi, 0, 1, 2, 3, 4, 5, 6, 7,
                                 8, 9, 10, 11, 12, 13, 14, 15);
}

static __device__ __forceinline__ v16bf frag_from_f32(const float* row, int kgrp) {
  const float* p0 = row + kgrp * 8;
  const float* p1 = row + 16 + kgrp * 8;
  v16bf r;
#pragma unroll
  for (int e = 0; e < 8; ++e) {
    r[e]     = (__bf16)p0[e];
    r[e + 8] = (__bf16)p1[e];
  }
  return r;
}

static __device__ __forceinline__ v8f wmma_bf16(v16bf a, v16bf b, v8f c) {
  // D(16x16 f32) = A(16x32 bf16) * B(32x16 bf16) + C
  return __builtin_amdgcn_wmma_f32_16x16x32_bf16(false, a, false, b,
                                                 (short)0, c, false, false);
}

// ---------------------------------------------------------------------------
// Stage 0: one-shot fp32 -> bf16 conversion of the four weight matrices.
// Removes per-tile re-conversion from the GEMM hot loops and halves the
// weight bytes pulled through L2 there. Grid: (1024 blocks, 4 matrices).
// ---------------------------------------------------------------------------
__global__ __launch_bounds__(256) void mha_cvt_weights(
    const float* __restrict__ wq, const float* __restrict__ wk,
    const float* __restrict__ wv, const float* __restrict__ dw,
    __bf16* __restrict__ wqb, __bf16* __restrict__ wkb,
    __bf16* __restrict__ wvb, __bf16* __restrict__ dwb) {
  const int m = blockIdx.y;
  const float* src = m == 0 ? wq : (m == 1 ? wk : (m == 2 ? wv : dw));
  __bf16*      dst = m == 0 ? wqb : (m == 1 ? wkb : (m == 2 ? wvb : dwb));
  const size_t i = ((size_t)blockIdx.x * 256 + threadIdx.x) * 4;
  const v4f f = *reinterpret_cast<const v4f*>(src + i);
  v8bf o;  // write 4 converted values (low half of an 8-vector store kept scalar)
  dst[i + 0] = (__bf16)f.x;
  dst[i + 1] = (__bf16)f.y;
  dst[i + 2] = (__bf16)f.z;
  dst[i + 3] = (__bf16)f.w;
  (void)o;
}

// ---------------------------------------------------------------------------
// Stage 1: qp/kp/vp = X @ W.T + b, activations fp32 -> bf16 on the fly,
// weights already bf16. Head-split outputs:
//   qh/kh: [b, h, s, d]   vhT: [b, h, d, s] (transposed so P@V B-frags are fast)
// Grid: (64 m-blocks of 64 rows, 8 n-blocks of 128 cols, 3 projections), 256 thr.
// Each wave owns a 16x64 output strip (A-fragment reused x4).
// ---------------------------------------------------------------------------
__global__ __launch_bounds__(256) void mha_qkv_proj(
    const float* __restrict__ q, const float* __restrict__ k,
    const float* __restrict__ v, const __bf16* __restrict__ wqb,
    const __bf16* __restrict__ wkb, const __bf16* __restrict__ wvb,
    const float* __restrict__ bq, const float* __restrict__ bk,
    const float* __restrict__ bv, __bf16* __restrict__ qh,
    __bf16* __restrict__ kh, __bf16* __restrict__ vhT) {
  const int lane = threadIdx.x & 31;
  const int wave = threadIdx.x >> 5;
  const int kgrp = lane >> 4;
  const int lcol = lane & 15;
  const int proj = blockIdx.z;  // 0=q 1=k 2=v

  const float*  x    = proj == 0 ? q   : (proj == 1 ? k   : v);
  const __bf16* w    = proj == 0 ? wqb : (proj == 1 ? wkb : wvb);
  const float*  bias = proj == 0 ? bq  : (proj == 1 ? bk  : bv);

  const int mbase = blockIdx.x * 64 + (wave & 3) * 16;
  const int nbase = blockIdx.y * 128 + (wave >> 2) * 64;

  const float*  arow = x + (size_t)(mbase + lcol) * MDIM;
  const __bf16* brow[4];
#pragma unroll
  for (int t = 0; t < 4; ++t)
    brow[t] = w + (size_t)(nbase + t * 16 + lcol) * MDIM;

  v8f c[4] = {};
  for (int kk = 0; kk < MDIM; kk += 32) {
    __builtin_prefetch(arow + kk + 256, 0, 3);       // global_prefetch_b8
    v16bf a = frag_from_f32(arow + kk, kgrp);
#pragma unroll
    for (int t = 0; t < 4; ++t)
      c[t] = wmma_bf16(a, frag_from_bf16(brow[t] + kk, kgrp), c[t]);
  }

#pragma unroll
  for (int t = 0; t < 4; ++t) {
    const int   col = nbase + t * 16 + lcol;
    const float bb  = bias[col];
    const int   h   = col >> 6, d = col & 63;
#pragma unroll
    for (int r = 0; r < 8; ++r) {
      const int   mrow = mbase + r + kgrp * 8;       // row in [0, B*S)
      const int   bi   = mrow >> 11, s = mrow & (SEQ - 1);
      const float val  = c[t][r] + bb;
      if (proj == 2) {
        vhT[(((size_t)bi * NHEADS + h) * DEPTHD + d) * SEQ + s] = (__bf16)val;
      } else {
        __bf16* dst = (proj == 0) ? qh : kh;
        dst[(((size_t)bi * NHEADS + h) * SEQ + s) * DEPTHD + d] = (__bf16)val;
      }
    }
  }
}

// ---------------------------------------------------------------------------
// Stage 2: per (b, h, 16-query-row tile):
//   1) logits(16x2048) = qh @ kh^T * 0.125 + mask*-1e9  -> LDS panel
//   2) row softmax in LDS (wave32 shuffle reductions), stream attn to HBM once
//   3) ctx(16x64) = P @ vh, K split across wave pairs, LDS combine, * 1/sum
// Grid: (128 qtiles, 16 heads, 2 batch), 256 thr, 131,392 B dynamic LDS.
// ---------------------------------------------------------------------------
__global__ __launch_bounds__(256) void mha_attention(
    const __bf16* __restrict__ qh, const __bf16* __restrict__ kh,
    const __bf16* __restrict__ vhT, const float* __restrict__ mask,
    float* __restrict__ attn, __bf16* __restrict__ ctxb) {
  extern __shared__ float smem[];
  float* P      = smem;               // 16 x PST fp32 logits / probs
  float* invRow = smem + 16 * PST;    // 16 reciprocal row sums

  const int lane = threadIdx.x & 31;
  const int wave = threadIdx.x >> 5;
  const int kgrp = lane >> 4;
  const int lcol = lane & 15;

  const int qtile = blockIdx.x;
  const int h     = blockIdx.y;
  const int b     = blockIdx.z;
  const size_t headrow = ((size_t)b * NHEADS + h) * SEQ;

  // Q fragments for this wave's 16 rows (depth 64 = two K=32 fragments)
  const __bf16* qp = qh + (headrow + qtile * 16 + lcol) * DEPTHD;
  const v16bf aq0 = frag_from_bf16(qp, kgrp);
  const v16bf aq1 = frag_from_bf16(qp + 32, kgrp);

  // ---- logits into LDS: wave w owns key tiles w, w+8, ... ----
  for (int kt = wave; kt < SEQ / 16; kt += 8) {
    const int     key = kt * 16 + lcol;
    const __bf16* kp  = kh + (headrow + key) * DEPTHD;
    __builtin_prefetch(kp + 128 * DEPTHD, 0, 3);  // this wave's next key tile
    v8f c = {};
    c = wmma_bf16(aq0, frag_from_bf16(kp, kgrp), c);
    c = wmma_bf16(aq1, frag_from_bf16(kp + 32, kgrp), c);
    const float mval = mask[b * SEQ + key] * -1e9f;
#pragma unroll
    for (int r = 0; r < 8; ++r)
      P[(r + kgrp * 8) * PST + key] = c[r] * 0.125f + mval;
  }
  __syncthreads();

  // ---- softmax: wave w handles rows 2w, 2w+1 ----
#pragma unroll
  for (int rr = 0; rr < 2; ++rr) {
    const int row  = wave * 2 + rr;
    float*    Prow = P + row * PST;
    float mx = -3.4e38f;
    for (int j = lane; j < SEQ; j += 32) mx = fmaxf(mx, Prow[j]);
#pragma unroll
    for (int off = 16; off; off >>= 1) mx = fmaxf(mx, __shfl_xor(mx, off, 32));
    float sum = 0.f;
    for (int j = lane; j < SEQ; j += 32) {
      const float e = __expf(Prow[j] - mx);
      Prow[j] = e;                       // keep unnormalized exp for P@V
      sum += e;
    }
#pragma unroll
    for (int off = 16; off; off >>= 1) sum += __shfl_xor(sum, off, 32);
    const float inv = 1.0f / sum;
    float* arow = attn + (headrow + qtile * 16 + row) * (size_t)SEQ;
    for (int j = lane; j < SEQ; j += 32) arow[j] = Prow[j] * inv;  // single HBM pass
    if (lane == 0) invRow[row] = inv;
  }
  __syncthreads();

  // ---- ctx = P @ vh : wave -> (ntile = w%4 of depth, khalf = w/4 of keys) ----
  const int     ntile = wave & 3;
  const int     khalf = wave >> 2;
  const int     dcol  = ntile * 16 + lcol;
  const __bf16* vrow  = vhT + (((size_t)b * NHEADS + h) * DEPTHD + dcol) * SEQ;

  v8f cc = {};
  for (int kt = 0; kt < 32; ++kt) {
    const int    k0   = khalf * 1024 + kt * 32;
    __builtin_prefetch(vrow + k0 + 512, 0, 3);
    const float* prow = P + lcol * PST + k0;   // A row = query row lcol
    v16bf a;
#pragma unroll
    for (int e = 0; e < 8; ++e) {
      a[e]     = (__bf16)prow[kgrp * 8 + e];
      a[e + 8] = (__bf16)prow[16 + kgrp * 8 + e];
    }
    cc = wmma_bf16(a, frag_from_bf16(vrow + k0, kgrp), cc);
  }
  __syncthreads();

  float* scr = P;  // panel no longer needed: reuse as 4x(16x16) combine scratch
  if (khalf == 1) {
#pragma unroll
    for (int r = 0; r < 8; ++r)
      scr[ntile * 256 + (r + kgrp * 8) * 16 + lcol] = cc[r];
  }
  __syncthreads();
  if (khalf == 0) {
#pragma unroll
    for (int r = 0; r < 8; ++r) {
      const int   m   = r + kgrp * 8;
      const float val = (cc[r] + scr[ntile * 256 + m * 16 + lcol]) * invRow[m];
      const int   s   = qtile * 16 + m;
      ctxb[((size_t)b * SEQ + s) * MDIM + h * DEPTHD + dcol] = (__bf16)val;
    }
  }
}

// ---------------------------------------------------------------------------
// Stage 3: out = ctx(bf16) @ dense_w(bf16).T + dense_b  (fp32 out)
// ---------------------------------------------------------------------------
__global__ __launch_bounds__(256) void mha_dense(
    const __bf16* __restrict__ ctxb, const __bf16* __restrict__ dwb,
    const float* __restrict__ db, float* __restrict__ out) {
  const int lane = threadIdx.x & 31;
  const int wave = threadIdx.x >> 5;
  const int kgrp = lane >> 4;
  const int lcol = lane & 15;

  const int mbase = blockIdx.x * 64 + (wave & 3) * 16;
  const int nbase = blockIdx.y * 128 + (wave >> 2) * 64;

  const __bf16* arow = ctxb + (size_t)(mbase + lcol) * MDIM;
  const __bf16* brow[4];
#pragma unroll
  for (int t = 0; t < 4; ++t)
    brow[t] = dwb + (size_t)(nbase + t * 16 + lcol) * MDIM;

  v8f c[4] = {};
  for (int kk = 0; kk < MDIM; kk += 32) {
    __builtin_prefetch(arow + kk + 256, 0, 3);
    v16bf a = frag_from_bf16(arow + kk, kgrp);
#pragma unroll
    for (int t = 0; t < 4; ++t)
      c[t] = wmma_bf16(a, frag_from_bf16(brow[t] + kk, kgrp), c[t]);
  }

#pragma unroll
  for (int t = 0; t < 4; ++t) {
    const int   col = nbase + t * 16 + lcol;
    const float bb  = db[col];
#pragma unroll
    for (int r = 0; r < 8; ++r) {
      const int mrow = mbase + r + kgrp * 8;
      out[(size_t)mrow * MDIM + col] = c[t][r] + bb;
    }
  }
}

// ---------------------------------------------------------------------------
extern "C" void kernel_launch(void* const* d_in, const int* in_sizes, int n_in,
                              void* d_out, int out_size, void* d_ws, size_t ws_size,
                              hipStream_t stream) {
  const float* v    = (const float*)d_in[0];
  const float* k    = (const float*)d_in[1];
  const float* q    = (const float*)d_in[2];
  const float* mask = (const float*)d_in[3];
  const float* wq   = (const float*)d_in[4];
  const float* bq   = (const float*)d_in[5];
  const float* wk   = (const float*)d_in[6];
  const float* bk   = (const float*)d_in[7];
  const float* wv   = (const float*)d_in[8];
  const float* bv   = (const float*)d_in[9];
  const float* dw   = (const float*)d_in[10];
  const float* db   = (const float*)d_in[11];

  float* out  = (float*)d_out;
  float* attn = out + (size_t)BATCH * SEQ * MDIM;  // tuple order: (out, attn)

  // Workspace (~42MB): 4 x 8MB bf16 activation regions + 4 x 2MB bf16 weights.
  const size_t headElems = (size_t)BATCH * NHEADS * SEQ * DEPTHD;  // 4,194,304
  const size_t wElems    = (size_t)MDIM * MDIM;                    // 1,048,576
  __bf16* qh   = (__bf16*)d_ws;
  __bf16* kh   = qh + headElems;
  __bf16* vhT  = kh + headElems;
  __bf16* ctxb = vhT + headElems;
  __bf16* wqb  = ctxb + headElems;
  __bf16* wkb  = wqb + wElems;
  __bf16* wvb  = wkb + wElems;
  __bf16* dwb  = wvb + wElems;

  const dim3 blk(256, 1, 1);
  mha_cvt_weights<<<dim3(MDIM * MDIM / (256 * 4), 4, 1), blk, 0, stream>>>(
      wq, wk, wv, dw, wqb, wkb, wvb, dwb);
  mha_qkv_proj<<<dim3(64, 8, 3), blk, 0, stream>>>(q, k, v, wqb, wkb, wvb,
                                                   bq, bk, bv, qh, kh, vhT);
  const size_t shbytes = (size_t)(16 * PST + 16) * sizeof(float);  // 131,392 B
  mha_attention<<<dim3(SEQ / 16, NHEADS, BATCH), blk, shbytes, stream>>>(
      qh, kh, vhT, mask, attn, ctxb);
  mha_dense<<<dim3(64, 8, 1), blk, 0, stream>>>(ctxb, dwb, db, out);
}
